// Model_6562710028433
// MI455X (gfx1250) — matrix operand order repaired and checked by reading, not compiled, not run on hardware
//
#include <hip/hip_runtime.h>
#include <hip/hip_bf16.h>
#include <cstdint>
#include <cstddef>

// ---------------------------------------------------------------------------
// BRITS-style imputation RNN for MI455X (gfx1250, wave32, WMMA).
//  - recurrent state h,c kept in registers for all 288 steps
//  - gate GEMM  gates^T = [W_hh | W_ih | bias] x [h_dec ; c_c,m ; 1]^T
//    via v_wmma_f32_16x16x32_bf16 (3 K-slices, fp32 accumulate)
//  - weights staged once into LDS, pre-swizzled into WMMA A-layout
//  - batch mapped to WMMA N so per-row scalar math is lane-local
//  - branchless transcendentals (v_exp_f32 / v_rcp_f32) so the recurrent
//    loop has no divergent EXEC regions between WMMAs
// ---------------------------------------------------------------------------

typedef __attribute__((ext_vector_type(16))) __bf16 v16bf;
typedef __attribute__((ext_vector_type(8)))  float  v8f;
typedef __attribute__((ext_vector_type(4)))  float  f32x4;
typedef __attribute__((ext_vector_type(2)))  float  f32x2;

union F8     { v8f v; float f[8]; };
union BF16V  { v16bf v; unsigned short u[16]; };

__device__ __forceinline__ unsigned short f2bf(float x) {
  unsigned int u = __builtin_bit_cast(unsigned int, x);
  u += 0x7fffu + ((u >> 16) & 1u);            // round-to-nearest-even
  return (unsigned short)(u >> 16);
}
__device__ __forceinline__ float fast_rcp(float x) { return __builtin_amdgcn_rcpf(x); }
__device__ __forceinline__ float sigmf(float x) {
  return fast_rcp(1.f + __expf(-x));          // branchless: v_exp + v_rcp
}
__device__ __forceinline__ float tanh_fast(float x) {
  return 1.f - 2.f * fast_rcp(__expf(2.f * x) + 1.f);  // branchless tanh
}
__device__ __forceinline__ float clipf(float v, float lo, float hi) {
  return fminf(fmaxf(v, lo), hi);
}

#define TT 288
#define BB 4096
#define NBLK 64     // 64 blocks * 4 waves * 16 rows = 4096

// ---------------------------------------------------------------------------
// Kernel 1: per-step mask denominators den[t] = sum_{b,f} masks[b,t,f]
// ---------------------------------------------------------------------------
__global__ __launch_bounds__(256)
void den_kernel(const float* __restrict__ masks, float* __restrict__ den) {
  int t = blockIdx.x;
  float s = 0.f;
  for (int b = threadIdx.x; b < BB; b += 256) {
    const float* p = masks + ((size_t)b * TT + t) * 2;
    s += p[0] + p[1];
  }
#pragma unroll
  for (int off = 16; off > 0; off >>= 1) s += __shfl_xor(s, off, 32);
  __shared__ float red[8];
  if ((threadIdx.x & 31) == 0) red[threadIdx.x >> 5] = s;
  __syncthreads();
  if (threadIdx.x == 0) {
    float tot = 0.f;
    for (int w = 0; w < 8; ++w) tot += red[w];
    den[t] = tot;
  }
}

// ---------------------------------------------------------------------------
// Kernel 2: main persistent RNN. Block = 128 threads (4 waves), wave = 16 rows.
// ---------------------------------------------------------------------------
__global__ __launch_bounds__(128)
void britsm_rnn_kernel(const float* __restrict__ nsr,   const float* __restrict__ obs,
                       const float* __restrict__ masks, const float* __restrict__ deltas,
                       const float* __restrict__ max_sr, const float* __restrict__ sr_mean,
                       const float* __restrict__ sr_std,
                       const float* __restrict__ W_gh,  const float* __restrict__ b_gh,
                       const float* __restrict__ W_gx,  const float* __restrict__ b_gx,
                       const float* __restrict__ W_hist, const float* __restrict__ b_hist,
                       const float* __restrict__ W_fr,  const float* __restrict__ b_fr,
                       const float* __restrict__ W_wc,  const float* __restrict__ b_wc,
                       const float* __restrict__ W_ih,  const float* __restrict__ W_hh,
                       const float* __restrict__ b_ih,  const float* __restrict__ b_hh,
                       float* __restrict__ dout, float* __restrict__ numws) {
  // --- LDS staging -----------------------------------------------------
  __shared__ __attribute__((aligned(32))) unsigned short sWhhA[16][2][32][16]; // 32 KB
  __shared__ __attribute__((aligned(32))) unsigned short sWihA[16][32][16];    // 16 KB (W_ih|bias slice)
  __shared__ __attribute__((aligned(32))) unsigned short sHB[4][2][32][16];    // 8 KB (per-wave h^T B-frags)
  __shared__ f32x4 sWgh[64];       // {w0, w1, bias, 0}
  __shared__ f32x2 sWhist[64];
  __shared__ f32x2 sWfrT[33];      // transposed masked feat_reg: sWfrT[c] = {W[0][c], W[1][c]}
  __shared__ float sMisc[32];

  const int tid = threadIdx.x;

  // A-layout (16-bit, 16x32): lane l<16: M=l, K = {0..7, 16..23}; lane>=16: K={8..15,24..31}
  for (int idx = tid; idx < 16 * 2 * 32 * 16; idx += 128) {
    int e = idx & 15, ln = (idx >> 4) & 31, kt = (idx >> 9) & 1, mt = idx >> 10;
    int M = ln & 15, gg = ln >> 4;
    int half = e >> 3, id8 = e & 7;
    int K = kt * 32 + half * 16 + gg * 8 + id8;
    sWhhA[mt][kt][ln][e] = f2bf(W_hh[(mt * 16 + M) * 64 + K]);
  }
  // third K-slice: columns [W_ih(4) | b_ih+b_hh (1) | zeros]
  for (int idx = tid; idx < 16 * 32 * 16; idx += 128) {
    int e = idx & 15, ln = (idx >> 4) & 31, mt = idx >> 9;
    int M = ln & 15, gg = ln >> 4;
    int half = e >> 3, id8 = e & 7;
    int K = half * 16 + gg * 8 + id8;
    int row = mt * 16 + M;
    float v = 0.f;
    if (K < 4)       v = W_ih[row * 4 + K];
    else if (K == 4) v = b_ih[row] + b_hh[row];
    sWihA[mt][ln][e] = f2bf(v);
  }
  for (int j = tid; j < 64; j += 128) {
    f32x4 w; w.x = W_gh[j * 2]; w.y = W_gh[j * 2 + 1]; w.z = b_gh[j]; w.w = 0.f;
    sWgh[j] = w;
    f32x2 h2; h2.x = W_hist[j]; h2.y = W_hist[64 + j];
    sWhist[j] = h2;
  }
  for (int c = tid; c < 33; c += 128) {
    f32x2 w;
    w.x = (c == 0) ? 0.f : W_fr[c];        // W_fr[0][0] masked
    w.y = (c == 1) ? 0.f : W_fr[33 + c];   // W_fr[1][1] masked
    sWfrT[c] = w;
  }
  if (tid == 0) {
    for (int f = 0; f < 2; ++f) {
      sMisc[0 + f]  = (1.5f * max_sr[f] - sr_mean[f]) / sr_std[f];  // ub
      sMisc[2 + f]  = (0.f - sr_mean[f]) / sr_std[f];               // lb
      sMisc[4 + f]  = b_hist[f];
      sMisc[6 + f]  = b_fr[f];
      sMisc[8 + f]  = W_gx[f * 2 + f];                               // diag of W_gx
      sMisc[10 + f] = b_gx[f];
      sMisc[20 + f] = b_wc[f];
    }
    for (int c = 0; c < 8; ++c) sMisc[12 + c] = W_wc[c];
  }
  __syncthreads();

  // --- per-wave mapping -------------------------------------------------
  const int wave = tid >> 5, lane = tid & 31;
  const int g = lane >> 4, bl = lane & 15;          // lane -> (K/M half, batch col)
  const int bGlob = blockIdx.x * 64 + wave * 16 + bl;

  const float* nsrP = nsr    + (size_t)bGlob * TT * 2;
  const float* mP   = masks  + (size_t)bGlob * TT * 2;
  const float* dP   = deltas + (size_t)bGlob * TT * 2;
  const float* oP   = obs    + (size_t)bGlob * TT * 31;
  float* impP = dout + 1 + (size_t)bGlob * TT * 2;

  const float ub0 = sMisc[0],  ub1 = sMisc[1],  lb0 = sMisc[2],  lb1 = sMisc[3];
  const float bh0 = sMisc[4],  bh1 = sMisc[5],  bfr0 = sMisc[6], bfr1 = sMisc[7];
  const float gxw0 = sMisc[8], gxw1 = sMisc[9], gxb0 = sMisc[10], gxb1 = sMisc[11];
  const float wwc0 = sMisc[12], wwc1 = sMisc[13], wwc2 = sMisc[14], wwc3 = sMisc[15];
  const float wwc4 = sMisc[16], wwc5 = sMisc[17], wwc6 = sMisc[18], wwc7 = sMisc[19];
  const float bwc0 = sMisc[20], bwc1 = sMisc[21];

  // recurrent state (fp32, C/D layout: tile hg, lane holds hidden j = hg*16+8*g+k)
  F8 hS[4], cS[4];
#pragma unroll
  for (int hg = 0; hg < 4; ++hg)
#pragma unroll
    for (int k = 0; k < 8; ++k) { hS[hg].f[k] = 0.f; cS[hg].f[k] = 0.f; }

  for (int t = 0; t < TT; ++t) {
    // scheduling barrier: keep per-step weight reads as LDS loads instead of
    // letting LICM hoist ~hundreds of them into VGPRs (spill avoidance)
    asm volatile("" ::: "memory");

    float x0 = nsrP[2 * t], x1 = nsrP[2 * t + 1];
    float m0 = mP[2 * t],   m1 = mP[2 * t + 1];
    float d0 = dP[2 * t],   d1 = dP[2 * t + 1];

    // temporal decay of h, x_h partials, scatter decayed h (bf16) into B-layout
    float xh0 = 0.f, xh1 = 0.f;
#pragma unroll
    for (int hg = 0; hg < 4; ++hg) {
#pragma unroll
      for (int k = 0; k < 8; ++k) {
        int j = hg * 16 + g * 8 + k;
        f32x4 wg = sWgh[j];
        float gh = __expf(-fmaxf(0.f, wg.x * d0 + wg.y * d1 + wg.z));
        float hd = hS[hg].f[k] * gh;
        hS[hg].f[k] = hd;
        f32x2 wh = sWhist[j];
        xh0 += hd * wh.x;
        xh1 += hd * wh.y;
        // B layout (bf16 32x16): lane l<16 holds K={0..7,16..23}, l>=16 K={8..15,24..31}, N=l&15
        sHB[wave][j >> 5][((j >> 3) & 1) * 16 + bl][((j >> 4) & 1) * 8 + (j & 7)] = f2bf(hd);
      }
    }
    xh0 += __shfl_xor(xh0, 16, 32);
    xh1 += __shfl_xor(xh1, 16, 32);
    xh0 = clipf(xh0 + bh0, lb0, ub0);
    xh1 = clipf(xh1 + bh1, lb1, ub1);

    float l0 = fabsf(x0 - xh0) * m0 + fabsf(x1 - xh1) * m1;
    float xc0 = m0 * x0 + (1.f - m0) * xh0;
    float xc1 = m1 * x1 + (1.f - m1) * xh1;

    // feat_reg over [x_c, x_obs] (masked diagonal already zeroed in LDS)
    f32x2 w0v = sWfrT[0], w1v = sWfrT[1];
    float z0 = w0v.x * xc0 + w1v.x * xc1 + bfr0;
    float z1 = w0v.y * xc0 + w1v.y * xc1 + bfr1;
    for (int kk = 0; kk < 31; ++kk) {
      float ov = oP[31 * t + kk];
      f32x2 wf = sWfrT[2 + kk];
      z0 += wf.x * ov;
      z1 += wf.y * ov;
    }
    z0 = clipf(z0, lb0, ub0);
    z1 = clipf(z1, lb1, ub1);
    float l1 = fabsf(x0 - z0) * m0 + fabsf(x1 - z1) * m1;

    float gx0 = __expf(-fmaxf(0.f, gxw0 * d0 + gxb0));
    float gx1 = __expf(-fmaxf(0.f, gxw1 * d1 + gxb1));
    float a0 = wwc0 * gx0 + wwc1 * gx1 + wwc2 * m0 + wwc3 * m1 + bwc0;
    float a1 = wwc4 * gx0 + wwc5 * gx1 + wwc6 * m0 + wwc7 * m1 + bwc1;
    float ch0 = clipf(a0 * z0 + (1.f - a0) * xh0, lb0, ub0);
    float ch1 = clipf(a1 * z1 + (1.f - a1) * xh1, lb1, ub1);
    float l2 = fabsf(x0 - ch0) * m0 + fabsf(x1 - ch1) * m1;

    float u0 = m0 * x0 + (1.f - m0) * ch0;   // c_c -> imputation + LSTM input
    float u1 = m1 * x1 + (1.f - m1) * ch1;
    if (lane < 16) { impP[2 * t] = u0; impP[2 * t + 1] = u1; }

    // per-step masked-L1 numerator partials (each row counted once: lanes < 16)
    float p0 = (lane < 16) ? l0 : 0.f;
    float p1 = (lane < 16) ? l1 : 0.f;
    float p2 = (lane < 16) ? l2 : 0.f;
#pragma unroll
    for (int off = 16; off > 0; off >>= 1) {
      p0 += __shfl_xor(p0, off, 32);
      p1 += __shfl_xor(p1, off, 32);
      p2 += __shfl_xor(p2, off, 32);
    }
    if (lane == 0) {
      float* nw = numws + ((size_t)(blockIdx.x * 4 + wave) * TT + t) * 3;
      nw[0] = p0; nw[1] = p1; nw[2] = p2;
    }

    // ---- B-fragment for the affine slice: rows [u0,u1,m0,m1,1,0,...] ----
    BF16V b2;
#pragma unroll
    for (int e = 0; e < 16; ++e) b2.u[e] = 0;
    if (g == 0) {
      b2.u[0] = f2bf(u0);
      b2.u[1] = f2bf(u1);
      b2.u[2] = f2bf(m0);
      b2.u[3] = f2bf(m1);
      b2.u[4] = 0x3F80;   // bf16(1.0)
    }

    // ---- gates^T = [W_hh | W_ih | bias] x [h_dec ; u ; 1]^T ----
    v16bf hb0 = *reinterpret_cast<const v16bf*>(&sHB[wave][0][lane][0]);
    v16bf hb1 = *reinterpret_cast<const v16bf*>(&sHB[wave][1][lane][0]);
#pragma unroll
    for (int hg = 0; hg < 4; ++hg) {
      F8 acc[4];   // q: 0=i, 1=f, 2=g, 3=o  (mt = q*4 + hg)
#pragma unroll
      for (int q = 0; q < 4; ++q) {
        int mt = q * 4 + hg;
        v16bf a2v = *reinterpret_cast<const v16bf*>(&sWihA[mt][lane][0]);
        v16bf a0v = *reinterpret_cast<const v16bf*>(&sWhhA[mt][0][lane][0]);
        v16bf a1v = *reinterpret_cast<const v16bf*>(&sWhhA[mt][1][lane][0]);
        v8f acz = (v8f)0.f;
        acz = __builtin_amdgcn_wmma_f32_16x16x32_bf16(
            false, a2v, false, b2.v, (short)0, acz, false, false);
        acz = __builtin_amdgcn_wmma_f32_16x16x32_bf16(
            false, a0v, false, hb0, (short)0, acz, false, false);
        acz = __builtin_amdgcn_wmma_f32_16x16x32_bf16(
            false, a1v, false, hb1, (short)0, acz, false, false);
        acc[q].v = acz;
      }
      // LSTM pointwise update for hidden group hg (branchless activations)
#pragma unroll
      for (int k = 0; k < 8; ++k) {
        float iv = sigmf(acc[0].f[k]);
        float fv = sigmf(acc[1].f[k]);
        float gv = tanh_fast(acc[2].f[k]);
        float ov = sigmf(acc[3].f[k]);
        float cn = fv * cS[hg].f[k] + iv * gv;
        cS[hg].f[k] = cn;
        hS[hg].f[k] = ov * tanh_fast(cn);
      }
    }
  }
}

// ---------------------------------------------------------------------------
// Kernel 3: deterministic final loss reduction: sum_t sum_p num/(den+eps)
// ---------------------------------------------------------------------------
__global__ __launch_bounds__(256)
void loss_kernel(const float* __restrict__ den, const float* __restrict__ numws,
                 float* __restrict__ out) {
  float acc = 0.f;
  for (int pid = threadIdx.x; pid < TT * 3; pid += 256) {
    int t = pid / 3, p = pid % 3;
    float s = 0.f;
    for (int w = 0; w < NBLK * 4; ++w) s += numws[((size_t)w * TT + t) * 3 + p];
    acc += s / (den[t] + 1e-5f);
  }
  __shared__ float red[256];
  red[threadIdx.x] = acc;
  __syncthreads();
  for (int off = 128; off > 0; off >>= 1) {
    if (threadIdx.x < off) red[threadIdx.x] += red[threadIdx.x + off];
    __syncthreads();
  }
  if (threadIdx.x == 0) out[0] = red[0];
}

// ---------------------------------------------------------------------------
extern "C" void kernel_launch(void* const* d_in, const int* in_sizes, int n_in,
                              void* d_out, int out_size, void* d_ws, size_t ws_size,
                              hipStream_t stream) {
  (void)in_sizes; (void)n_in; (void)out_size; (void)ws_size;
  const float* nsr     = (const float*)d_in[0];
  const float* obs     = (const float*)d_in[1];
  const float* masks   = (const float*)d_in[2];
  const float* deltas  = (const float*)d_in[3];
  const float* max_sr  = (const float*)d_in[4];
  const float* sr_mean = (const float*)d_in[5];
  const float* sr_std  = (const float*)d_in[6];
  const float* W_gh    = (const float*)d_in[7];
  const float* b_gh    = (const float*)d_in[8];
  const float* W_gx    = (const float*)d_in[9];
  const float* b_gx    = (const float*)d_in[10];
  const float* W_hist  = (const float*)d_in[11];
  const float* b_hist  = (const float*)d_in[12];
  const float* W_fr    = (const float*)d_in[13];
  const float* b_fr    = (const float*)d_in[14];
  const float* W_wc    = (const float*)d_in[15];
  const float* b_wc    = (const float*)d_in[16];
  const float* W_ih    = (const float*)d_in[17];
  const float* W_hh    = (const float*)d_in[18];
  const float* b_ih    = (const float*)d_in[19];
  const float* b_hh    = (const float*)d_in[20];

  float* out   = (float*)d_out;
  float* wsf   = (float*)d_ws;
  float* den   = wsf;          // 288 floats
  float* numws = wsf + 512;    // 256 wave-slots * 288 * 3 floats

  den_kernel<<<TT, 256, 0, stream>>>(masks, den);
  britsm_rnn_kernel<<<NBLK, 128, 0, stream>>>(
      nsr, obs, masks, deltas, max_sr, sr_mean, sr_std,
      W_gh, b_gh, W_gx, b_gx, W_hist, b_hist, W_fr, b_fr, W_wc, b_wc,
      W_ih, W_hh, b_ih, b_hh, out, numws);
  loss_kernel<<<1, 256, 0, stream>>>(den, numws, out);
}
